// LightweightGaborGuidedTransformer_75539884802829
// MI455X (gfx1250) — matrix operand order repaired and checked by reading, not compiled, not run on hardware
//
#include <hip/hip_runtime.h>
#include <hip/hip_bf16.h>

typedef __attribute__((ext_vector_type(16))) _Float16 v16h;
typedef __attribute__((ext_vector_type(8)))  float    v8f;

#define PIF 3.14159265358979323846f

static constexpr int NORI = 6;
static constexpr int WSZ  = 8;
static constexpr int E    = 32;
static constexpr int LSEQ = 64;
static constexpr int IMG  = 512;

// ---- LDS arena offsets (bytes); buffers are reused across pipeline stages ----
static constexpr int OFF_SQ = 0;        // f16 [64][32]  (q sequence)
static constexpr int OFF_SK = 4096;     // f16 [64][32]  (kv sequence)
static constexpr int OFF_W  = 8192;     // f16 weight slot, 8 KB (wq|wk|wv, then ow, m1w, m2w)
static constexpr int OFF_Q  = 16384;    // f16 [64][32]
static constexpr int OFF_K  = 20480;    // f16 [64][32]
static constexpr int OFF_V  = 24576;    // f16 [64][32]
static constexpr int OFF_P  = 28672;    // f16 [64][64]  per-head probs
static constexpr int OFF_AO = 36864;    // f16 [64][32]
static constexpr int OFF_X1 = 16384;    // f16 [64][32]  (reuses Q after attention)
static constexpr int OFF_HM = 20480;    // f16 [64][128] (reuses K/V/P)
static constexpr int OFF_X2 = 40960;    // f32 [64][32]
static constexpr int ARENA_BYTES = 49152;

// ---------------- WMMA fragment helpers (CDNA5 wave32 layouts) ----------------
// A 16x32 f16: lane -> M = lane&15, g = lane>>4; halves j=0..15 -> K=16g+j
__device__ __forceinline__ v16h load_fragA(const _Float16* base, int m0, int k0, int ld) {
  int lane = threadIdx.x & 31;
  const _Float16* p = base + (m0 + (lane & 15)) * ld + k0;
  int g16 = (lane >> 4) * 16;
  v16h a;
#pragma unroll
  for (int j = 0; j < 8; ++j) a[j] = p[g16 + j];
#pragma unroll
  for (int j = 0; j < 8; ++j) a[8 + j] = p[g16 + 8 + j];
  return a;
}

// B(k,n) = Wrow[n*ld + k]  (for D = A * W^T, W row-major [n][k])
// lanes 0-15: K=0..15, lanes 16-31: K=16..31 (within this 32-K step)
__device__ __forceinline__ v16h load_fragB_wt(const _Float16* Wrow, int n0, int k0, int ld) {
  int lane = threadIdx.x & 31;
  const _Float16* p = Wrow + (n0 + (lane & 15)) * ld + k0 + ((lane >> 4) * 16);
  v16h b;
#pragma unroll
  for (int j = 0; j < 16; ++j) b[j] = p[j];
  return b;
}

// B(k,n) = M[k*ld + n]  (k-major matrix, e.g. V for AO = P*V)
__device__ __forceinline__ v16h load_fragB_km(const _Float16* Mkn, int n0, int k0, int ld) {
  int lane = threadIdx.x & 31;
  int n = n0 + (lane & 15);
  int kb = k0 + ((lane >> 4) * 16);
  v16h b;
#pragma unroll
  for (int j = 0; j < 16; ++j) b[j] = Mkn[(kb + j) * ld + n];
  return b;
}

// D/C 16x16 f32: lane -> N = lane&15; VGPR r -> M = r + 8*(lane>>4)
__device__ __forceinline__ void store_fragD_f16(_Float16* dst, v8f d, int m0, int n0, int ld) {
  int lane = threadIdx.x & 31;
  int n = n0 + (lane & 15), g = lane >> 4;
#pragma unroll
  for (int r = 0; r < 8; ++r) dst[(m0 + r + 8 * g) * ld + n] = (_Float16)d[r];
}

__device__ __forceinline__ v8f bias_fragC(const float* bias, int n0) {
  int lane = threadIdx.x & 31;
  float bv = bias[n0 + (lane & 15)];
  v8f c;
#pragma unroll
  for (int r = 0; r < 8; ++r) c[r] = bv;
  return c;
}

__device__ __forceinline__ v8f wmma16(v16h a, v16h b, v8f c) {
  return __builtin_amdgcn_wmma_f32_16x16x32_f16(false, a, false, b, (short)0, c, false, false);
}

// ---------------------------- fused kernel ----------------------------
__global__ __launch_bounds__(256) void gabor_window_transformer(
    const float* __restrict__ x,
    const float* __restrict__ sigma, const float* __restrict__ lambd,
    const float* __restrict__ psi,   const float* __restrict__ gamma_,
    const float* __restrict__ theta, const float* __restrict__ dir_w,
    const float* __restrict__ qw,  const float* __restrict__ qb,
    const float* __restrict__ kvw, const float* __restrict__ kvb,
    const float* __restrict__ in_w, const float* __restrict__ in_b,
    const float* __restrict__ ow,  const float* __restrict__ ob,
    const float* __restrict__ m1w, const float* __restrict__ m1b,
    const float* __restrict__ m2w, const float* __restrict__ m2b,
    const float* __restrict__ pw,  const float* __restrict__ pb,
    float* __restrict__ out, int nwx, int nwy)
{
  __shared__ __align__(16) unsigned char arena[ARENA_BYTES];
  __shared__ float patch[12 * 12];
  __shared__ float sKr[25], sKi[25];
  __shared__ float sBase[64], sDetail[64];

  const int tid  = threadIdx.x;
  const int wv   = tid >> 5;
  const int lane = tid & 31;

  const int win = blockIdx.x;
  const int b   = win / (nwy * nwx);
  const int wy  = (win / nwx) % nwy;
  const int wx  = win % nwx;
  const int y0  = wy * WSZ, x0 = wx * WSZ;

  // ---- Stage 0: combined Gabor kernels (softmax(dir_w)-weighted) ----
  if (tid < 25) { sKr[tid] = 0.f; sKi[tid] = 0.f; }
  if (tid < 144) {
    int py = tid / 12, px = tid % 12;
    int iy = y0 + py - 2, ix = x0 + px - 2;
    float v = 0.f;
    if (iy >= 0 && iy < IMG && ix >= 0 && ix < IMG)
      v = x[((size_t)b * IMG + iy) * IMG + ix];
    patch[tid] = v;
  }
  __syncthreads();
  if (tid < NORI * 25) {
    int o = tid / 25, p = tid % 25;
    float mx = dir_w[0];
#pragma unroll
    for (int i = 1; i < NORI; ++i) mx = fmaxf(mx, dir_w[i]);
    float den = 0.f;
#pragma unroll
    for (int i = 0; i < NORI; ++i) den += __expf(dir_w[i] - mx);
    float wo = __expf(dir_w[o] - mx) / den;
    float s  = log1pf(__expf(sigma[o]))  + 0.1f;
    float l  = log1pf(__expf(lambd[o]))  + 0.1f;
    float g  = log1pf(__expf(gamma_[o])) + 0.1f;
    float th = tanhf(theta[o]) * PIF;
    float ps = tanhf(psi[o])   * PIF;
    int py = p / 5, px = p % 5;
    float yy = (py - 2) * 0.5f, xx = (px - 2) * 0.5f;
    float ct = __cosf(th), st = __sinf(th);
    float xt =  xx * ct + yy * st;
    float yt = -xx * st + yy * ct;
    float env   = __expf(-(xt * xt + g * g * yt * yt) / (2.f * s * s));
    float phase = 2.f * PIF * xt / l + ps;
    atomicAdd(&sKr[p], wo * env * __cosf(phase));   // ds_add_f32
    atomicAdd(&sKi[p], wo * env * __sinf(phase));
  }
  __syncthreads();

  // ---- Stage 1: 5x5 SAME conv -> base (real) / detail (imag) ----
  if (tid < 64) {
    int ly = tid >> 3, lx = tid & 7;
    float ar = 0.f, ai = 0.f;
#pragma unroll
    for (int dy = 0; dy < 5; ++dy)
#pragma unroll
      for (int dx = 0; dx < 5; ++dx) {
        float pv = patch[(ly + dy) * 12 + (lx + dx)];
        ar += sKr[dy * 5 + dx] * pv;
        ai += sKi[dy * 5 + dx] * pv;
      }
    sBase[tid] = ar; sDetail[tid] = ai;
  }
  __syncthreads();

  // ---- Stage 2: lift to E channels (f16) + stage QKV weights ----
  _Float16* SQ = (_Float16*)(arena + OFF_SQ);
  _Float16* SK = (_Float16*)(arena + OFF_SK);
  _Float16* Wh = (_Float16*)(arena + OFF_W);
  for (int i = tid; i < LSEQ * E; i += 256) {
    int l = i >> 5, e = i & 31;
    SQ[i] = (_Float16)(qw[e] * sDetail[l] + qb[e]);
    SK[i] = (_Float16)(kvw[e] * sBase[l] + kvb[e]);
  }
  for (int i = tid; i < 3 * E * E; i += 256) Wh[i] = (_Float16)in_w[i];
  __syncthreads();

  // ---- Stage 3: Q/K/V projections (24 WMMA tiles over 8 waves) ----
  _Float16* Qh = (_Float16*)(arena + OFF_Q);
  _Float16* Kh = (_Float16*)(arena + OFF_K);
  _Float16* Vh = (_Float16*)(arena + OFF_V);
  for (int t = 3 * wv; t < 3 * wv + 3; ++t) {
    int mat = t >> 3, r = t & 7, mt = r >> 1, nt = r & 1;
    const _Float16* A = (mat == 0) ? SQ : SK;
    _Float16* D = (_Float16*)(arena + OFF_Q + mat * 4096);
    v16h af = load_fragA(A, 16 * mt, 0, E);
    v16h bf = load_fragB_wt(Wh + mat * (E * E), 16 * nt, 0, E);
    v8f  c  = bias_fragC(in_b + mat * E, 16 * nt);
    c = wmma16(af, bf, c);
    store_fragD_f16(D, c, 16 * mt, 16 * nt, E);
  }
  __syncthreads();

  // ---- Stage 4/5: per-head scores + softmax + AV (heads sequential) ----
  _Float16* Ph = (_Float16*)(arena + OFF_P);
  _Float16* AO = (_Float16*)(arena + OFF_AO);
  const float scale = 0.3535533905932738f; // 1/sqrt(8)
  for (int h = 0; h < 4; ++h) {
    if (wv < 4) { // scores: wave = M-tile, all 4 N-tiles -> full rows in-wave
      int m0 = 16 * wv;
      int nl = lane & 15, gq = lane >> 4;
      bool gmatch = (gq == (h >> 1));
      int jlo = 8 * (h & 1);
      v16h af = load_fragA(Qh, m0, 0, E);
      v8f d[4];
#pragma unroll
      for (int nt = 0; nt < 4; ++nt) {
        const _Float16* p = Kh + (16 * nt + nl) * E + 16 * gq;
        v16h bf; // K^T restricted to head h's 8 channels, zero elsewhere
#pragma unroll
        for (int j = 0; j < 16; ++j)
          bf[j] = (gmatch && j >= jlo && j < jlo + 8) ? p[j] : (_Float16)0.f;
        v8f z = {};
        d[nt] = wmma16(af, bf, z);
      }
#pragma unroll
      for (int r = 0; r < 8; ++r) {
        float v0 = d[0][r] * scale, v1 = d[1][r] * scale;
        float v2 = d[2][r] * scale, v3 = d[3][r] * scale;
        float mx = fmaxf(fmaxf(v0, v1), fmaxf(v2, v3));
#pragma unroll
        for (int m = 1; m < 16; m <<= 1) mx = fmaxf(mx, __shfl_xor(mx, m, 32));
        float e0 = __expf(v0 - mx), e1 = __expf(v1 - mx);
        float e2 = __expf(v2 - mx), e3 = __expf(v3 - mx);
        float sm = e0 + e1 + e2 + e3;
#pragma unroll
        for (int m = 1; m < 16; m <<= 1) sm += __shfl_xor(sm, m, 32);
        float inv = 1.f / sm;
        int row = m0 + r + 8 * gq;
        Ph[row * 64 +  0 + nl] = (_Float16)(e0 * inv);
        Ph[row * 64 + 16 + nl] = (_Float16)(e1 * inv);
        Ph[row * 64 + 32 + nl] = (_Float16)(e2 * inv);
        Ph[row * 64 + 48 + nl] = (_Float16)(e3 * inv);
      }
    }
    __syncthreads();
    if (wv < 4) { // AO[:, head cols] = P_h @ V[:, head cols]
      int m0 = 16 * wv;
      int n0 = 16 * (h >> 1);
      v8f c = {};
#pragma unroll
      for (int ks = 0; ks < 2; ++ks) {
        v16h af = load_fragA(Ph, m0, 32 * ks, 64);
        v16h bf = load_fragB_km(Vh, n0, 32 * ks, E);
        c = wmma16(af, bf, c);
      }
      int nl = lane & 15, gq = lane >> 4;
      int lo = 8 * (h & 1);
      if (nl >= lo && nl < lo + 8) {
#pragma unroll
        for (int r = 0; r < 8; ++r)
          AO[(m0 + r + 8 * gq) * E + n0 + nl] = (_Float16)c[r];
      }
    }
    __syncthreads();
  }

  // ---- Stage 6: output projection X1 = AO @ ow^T + ob ----
  for (int i = tid; i < E * E; i += 256) Wh[i] = (_Float16)ow[i];
  __syncthreads();
  _Float16* X1 = (_Float16*)(arena + OFF_X1);
  {
    int mt = wv >> 1, nt = wv & 1;
    v16h af = load_fragA(AO, 16 * mt, 0, E);
    v16h bf = load_fragB_wt(Wh, 16 * nt, 0, E);
    v8f  c  = bias_fragC(ob, 16 * nt);
    c = wmma16(af, bf, c);
    store_fragD_f16(X1, c, 16 * mt, 16 * nt, E);
  }
  __syncthreads();

  // ---- Stage 7: Hmid = relu(X1 @ m1w^T + m1b)  (64x128) ----
  for (int i = tid; i < 128 * E; i += 256) Wh[i] = (_Float16)m1w[i];
  __syncthreads();
  _Float16* HM = (_Float16*)(arena + OFF_HM);
  {
    int nt = wv;
    v16h bf = load_fragB_wt(Wh, 16 * nt, 0, E);
    v8f  cb = bias_fragC(m1b, 16 * nt);
    int nl = lane & 15, gq = lane >> 4;
#pragma unroll
    for (int mt = 0; mt < 4; ++mt) {
      v16h af = load_fragA(X1, 16 * mt, 0, E);
      v8f  c  = wmma16(af, bf, cb);
#pragma unroll
      for (int r = 0; r < 8; ++r)
        HM[(16 * mt + r + 8 * gq) * 128 + 16 * nt + nl] = (_Float16)fmaxf(c[r], 0.f);
    }
  }
  __syncthreads();

  // ---- Stage 8: X2 = X1 + Hmid @ m2w^T + m2b  (64x32, f32) ----
  for (int i = tid; i < E * 128; i += 256) Wh[i] = (_Float16)m2w[i];
  __syncthreads();
  float* X2 = (float*)(arena + OFF_X2);
  {
    int mt = wv >> 1, nt = wv & 1;
    int nl = lane & 15, gq = lane >> 4;
    int n = 16 * nt + nl;
    float bm = m2b[n];
    v8f c;
#pragma unroll
    for (int r = 0; r < 8; ++r)
      c[r] = (float)X1[(16 * mt + r + 8 * gq) * E + n] + bm;
#pragma unroll
    for (int ks = 0; ks < 4; ++ks) {
      v16h af = load_fragA(HM, 16 * mt, 32 * ks, 128);
      v16h bf = load_fragB_wt(Wh, 16 * nt, 32 * ks, 128);
      c = wmma16(af, bf, c);
    }
#pragma unroll
    for (int r = 0; r < 8; ++r)
      X2[(16 * mt + r + 8 * gq) * E + n] = c[r];
  }
  __syncthreads();

  // ---- Stage 9: out = X2 @ pw^T + pb + detail ----
  if (tid < 64) {
    int ly = tid >> 3, lx = tid & 7;
    float acc = pb[0] + sDetail[tid];
#pragma unroll
    for (int e = 0; e < E; ++e) acc += X2[tid * E + e] * pw[e];
    out[((size_t)b * IMG + (y0 + ly)) * IMG + (x0 + lx)] = acc;
  }
}

extern "C" void kernel_launch(void* const* d_in, const int* in_sizes, int n_in,
                              void* d_out, int out_size, void* d_ws, size_t ws_size,
                              hipStream_t stream) {
  (void)n_in; (void)out_size; (void)d_ws; (void)ws_size;
  const float* x      = (const float*)d_in[0];
  const float* sigma  = (const float*)d_in[1];
  const float* lambd  = (const float*)d_in[2];
  const float* psi    = (const float*)d_in[3];
  const float* gamma_ = (const float*)d_in[4];
  const float* theta  = (const float*)d_in[5];
  const float* dir_w  = (const float*)d_in[6];
  const float* qw     = (const float*)d_in[7];
  const float* qb     = (const float*)d_in[8];
  const float* kvw    = (const float*)d_in[9];
  const float* kvb    = (const float*)d_in[10];
  const float* in_w   = (const float*)d_in[11];
  const float* in_b   = (const float*)d_in[12];
  const float* ow     = (const float*)d_in[13];
  const float* ob     = (const float*)d_in[14];
  const float* m1w    = (const float*)d_in[15];
  const float* m1b    = (const float*)d_in[16];
  const float* m2w    = (const float*)d_in[17];
  const float* m2b    = (const float*)d_in[18];
  const float* pw     = (const float*)d_in[19];
  const float* pb     = (const float*)d_in[20];
  float* out = (float*)d_out;

  const int B   = in_sizes[0] / (IMG * IMG);
  const int nwx = IMG / WSZ, nwy = IMG / WSZ;
  const int nblocks = B * nwy * nwx;

  gabor_window_transformer<<<nblocks, 256, 0, stream>>>(
      x, sigma, lambd, psi, gamma_, theta, dir_w, qw, qb, kvw, kvb,
      in_w, in_b, ow, ob, m1w, m1b, m2w, m2b, pw, pb, out, nwx, nwy);
}
